// RNN_65068754534454
// MI455X (gfx1250) — matrix operand order repaired and checked by reading, not compiled, not run on hardware
//
#include <hip/hip_runtime.h>

#define IN   256
#define HID  512
#define OUT  256
#define BB   64
#define TT   1024
#define WIH_K (IN + HID)   // 768
#define NWG  16

typedef __attribute__((ext_vector_type(16))) __bf16 v16bf;
typedef __attribute__((ext_vector_type(8)))  float  v8f;

// Workspace layout (bytes)
#define WS_BAR    0
#define WS_WIHBF  256                              // 512*768 bf16 = 786432
#define WS_HST    (WS_WIHBF + HID * WIH_K * 2)     // 786688 ; 2 x 64x512 bf16
#define WS_RST    (WS_HST + 2 * BB * HID * 2)      // 917760 ; 2 x 64x256 bf16
// total ~983 KB

// ---------------------------------------------------------------------------
// WMMA helper: D = A(16x32 bf16) x B(32x16 bf16) + C(16x16 f32)
// ---------------------------------------------------------------------------
__device__ inline v8f wmma_bf16(v16bf a, v16bf b, v8f c) {
  return __builtin_amdgcn_wmma_f32_16x16x32_bf16(
      /*neg_a=*/false, a, /*neg_b=*/false, b,
      /*c_mod=*/(short)0, c, /*reuse_a=*/false, /*reuse_b=*/false);
}

// A tile loader from f32 source (per-lane), 16-bit A 16x32 layout (ISA 7.12.2):
//   lane = half*16 + m ; VGPR v0..3 = K(half*8 + 0..7), v4..7 = K(16 + half*8 + 0..7)
__device__ inline v16bf make_a_f32(const float* rowp, int half) {
  const float4* p0 = (const float4*)(rowp + half * 8);
  const float4* p1 = (const float4*)(rowp + 16 + half * 8);
  float4 f0 = p0[0], f1 = p0[1], f2 = p1[0], f3 = p1[1];
  v16bf a;
  a[0]=(__bf16)f0.x; a[1]=(__bf16)f0.y; a[2]=(__bf16)f0.z; a[3]=(__bf16)f0.w;
  a[4]=(__bf16)f1.x; a[5]=(__bf16)f1.y; a[6]=(__bf16)f1.z; a[7]=(__bf16)f1.w;
  a[8]=(__bf16)f2.x; a[9]=(__bf16)f2.y; a[10]=(__bf16)f2.z; a[11]=(__bf16)f2.w;
  a[12]=(__bf16)f3.x; a[13]=(__bf16)f3.y; a[14]=(__bf16)f3.z; a[15]=(__bf16)f3.w;
  return a;
}

// A tile loader from pre-converted bf16 state: two 16B loads, no conversion.
__device__ inline v16bf make_a_bf(const __bf16* rowp, int half) {
  const uint4* p0 = (const uint4*)(rowp + half * 8);
  const uint4* p1 = (const uint4*)(rowp + 16 + half * 8);
  union { uint4 u[2]; v16bf v; } cv;
  cv.u[0] = p0[0];
  cv.u[1] = p1[0];
  return cv.v;
}

// B tile loader (per-lane), 16-bit B 32x16 layout:
//   lane = half*16 + n ; VGPR v0..3 = K(half*8 + 0..7), v4..7 = K(16 + half*8 + 0..7)
__device__ inline v16bf make_b(const __bf16* colp, int half) {
  const uint4* p0 = (const uint4*)(colp + half * 8);
  const uint4* p1 = (const uint4*)(colp + 16 + half * 8);
  union { uint4 u[2]; v16bf v; } cv;
  cv.u[0] = p0[0];
  cv.u[1] = p1[0];
  return cv.v;
}

// ---------------------------------------------------------------------------
// Setup: W_ih -> bf16; seed f32 outputs at t=0 and bf16 state buffers (par 0)
// ---------------------------------------------------------------------------
__global__ void setup_kernel(const float* __restrict__ wih, __bf16* __restrict__ wih_bf,
                             const float* __restrict__ h0, const float* __restrict__ r0,
                             float* __restrict__ rout, float* __restrict__ hout,
                             __bf16* __restrict__ hst, __bf16* __restrict__ rst) {
  int i = blockIdx.x * blockDim.x + threadIdx.x;
  if (i < HID * WIH_K) wih_bf[i] = (__bf16)wih[i];
  if (i < BB * HID) {
    int b = i / HID, k = i % HID;
    hout[(size_t)b * 1025 * HID + k] = h0[i];
    hst[i] = (__bf16)h0[i];                    // parity-0 buffer, [b][k] packed
  }
  if (i < BB * OUT) {
    int b = i / OUT, o = i % OUT;
    rout[(size_t)b * 1025 * OUT + o] = r0[i];
    rst[i] = (__bf16)r0[i];
  }
}

// ---------------------------------------------------------------------------
// XW precompute: XW[(b,t), n] = b_ih[n] + sum_k x[b,t,k] * W_ih[n][k]
// written to h_out[b, t+1, n]  (consumed in-place as C-init by rnn_kernel).
// ---------------------------------------------------------------------------
__global__ void xw_kernel(const float* __restrict__ x, const __bf16* __restrict__ wih_bf,
                          const float* __restrict__ b_ih, float* __restrict__ hout) {
  int w    = blockIdx.x * (blockDim.x >> 5) + (threadIdx.x >> 5);
  int lane = threadIdx.x & 31;
  int half = lane >> 4, lm = lane & 15;
  int nt = w & 31;     // 32 N tiles over HID
  int mt = w >> 5;     // 4096 M tiles over B*T rows
  int n  = nt * 16 + lm;

  v8f c;
  float bias = b_ih[n];
#pragma unroll
  for (int i = 0; i < 8; ++i) c[i] = bias;

  const float*  arow = x + (size_t)(mt * 16 + lm) * IN;   // row = b*1024 + t
  const __bf16* bcol = wih_bf + (size_t)n * WIH_K;        // W_ih row n (Wx column n)
#pragma unroll
  for (int kc = 0; kc < IN / 32; ++kc) {
    v16bf a = make_a_f32(arow + kc * 32, half);
    v16bf b = make_b(bcol + kc * 32, half);
    c = wmma_bf16(a, b, c);
  }
#pragma unroll
  for (int i = 0; i < 8; ++i) {
    int row = mt * 16 + i + half * 8;      // global (b,t) row
    int bb = row >> 10, t = row & 1023;
    hout[((size_t)bb * 1025 + (t + 1)) * HID + n] = c[i];
  }
}

// ---------------------------------------------------------------------------
// Grid barrier (monotonic counter, reset via memset each launch)
// ---------------------------------------------------------------------------
__device__ inline void grid_barrier(unsigned* bar, unsigned target) {
  __syncthreads();
  if (threadIdx.x == 0) {
    __threadfence();
    atomicAdd(bar, 1u);
    while (__hip_atomic_load(bar, __ATOMIC_RELAXED, __HIP_MEMORY_SCOPE_AGENT) < target) {
      __builtin_amdgcn_s_sleep(1);
    }
    __threadfence();
  }
  __syncthreads();
}

// ---------------------------------------------------------------------------
// Persistent recurrent kernel: 16 WGs x 256 threads, t-loop, 1 barrier/step.
// State carried as bf16 ping-pong mirrors (L2-resident); f32 copies are the
// final outputs. Weights live in LDS (bf16) for all 1024 steps.
// ---------------------------------------------------------------------------
#define WHS_STRIDE (HID + 8)     // 520 bf16 (16B-aligned rows)
#define WOS_STRIDE (WIH_K + 8)   // 776 bf16

__global__ void __launch_bounds__(256) rnn_kernel(
    const float* __restrict__ wih, const float* __restrict__ who,
    const float* __restrict__ b_ho, float* __restrict__ rout,
    float* __restrict__ hout, __bf16* __restrict__ hst,
    __bf16* __restrict__ rst, unsigned* __restrict__ bar) {
  __shared__ __bf16 whs[32 * WHS_STRIDE];   // Wh columns g*32..g*32+31, K=512
  __shared__ __bf16 wos[16 * WOS_STRIDE];   // W_ho rows g*16..g*16+15, K=768
  __shared__ float  part[4][16][16];        // r-phase split-K partials

  const int g = blockIdx.x;
  const int tid = threadIdx.x;
  const int w = tid >> 5, lane = tid & 31;
  const int half = lane >> 4, lm = lane & 15;

  // ---- stage weights into LDS (f32 -> bf16), once ----
  for (int idx = tid; idx < 32 * HID; idx += 256) {
    int c = idx >> 9, k = idx & 511;
    whs[c * WHS_STRIDE + k] = (__bf16)wih[(size_t)(g * 32 + c) * WIH_K + IN + k];
  }
  for (int idx = tid; idx < 16 * WIH_K; idx += 256) {
    int o = idx / WIH_K, k = idx % WIH_K;
    wos[o * WOS_STRIDE + k] = (__bf16)who[(size_t)(g * 16 + o) * WIH_K + k];
  }
  __syncthreads();

  // ---- per-wave assignments ----
  const int h_mt = w & 3;                 // M tile over batch (4 x 16 rows)
  const int h_nc = w >> 2;                // 0..1 -> 16-col group within WG slice
  const int h_ncol_local = h_nc * 16 + lm;
  const size_t h_nbase_col = (size_t)(g * 32) + h_ncol_local;

  const int r_mt = w & 3;
  const bool r_hi = (w >= 4);             // K-chunks 12..23 vs 0..11
  const int r_o = g * 16 + lm;
  const float bho = b_ho[r_o];

  for (int t = 0; t < TT; ++t) {
    const int pcur = t & 1, pnxt = (t + 1) & 1;
    // ================= h phase =================
    {
      v8f c;
      // C init = XW tile (bias folded) living at h_out[:, t+1, :]
#pragma unroll
      for (int i = 0; i < 8; ++i) {
        int brow = h_mt * 16 + i + half * 8;
        c[i] = hout[((size_t)brow * 1025 + (t + 1)) * HID + h_nbase_col];
      }
      // prefetch next step's XW tile (HBM-resident stream) one step ahead
      if (t + 2 <= TT) {
        const float* pf = hout + ((size_t)(h_mt * 16 + lm) * 1025 + (t + 2)) * HID
                          + (size_t)(g * 32 + h_nc * 16);
        __builtin_prefetch(pf, 0, 1);
      }
      const __bf16* arow = hst + (size_t)pcur * BB * HID + (size_t)(h_mt * 16 + lm) * HID;
      const __bf16* bcol = whs + (size_t)h_ncol_local * WHS_STRIDE;
#pragma unroll 4
      for (int kc = 0; kc < HID / 32; ++kc) {
        v16bf a = make_a_bf(arow + kc * 32, half);
        v16bf b = make_b(bcol + kc * 32, half);
        c = wmma_bf16(a, b, c);
      }
#pragma unroll
      for (int i = 0; i < 8; ++i) {
        int brow = h_mt * 16 + i + half * 8;
        float v = c[i];
        v = v > 0.f ? v : 0.f;
        hout[((size_t)brow * 1025 + (t + 1)) * HID + h_nbase_col] = v;
        hst[(size_t)pnxt * BB * HID + (size_t)brow * HID + h_nbase_col] = (__bf16)v;
      }
    }

    grid_barrier(bar, (unsigned)(NWG * (t + 1)));

    // ================= r phase =================
    {
      v8f c;
#pragma unroll
      for (int i = 0; i < 8; ++i) c[i] = r_hi ? 0.f : bho;

      const __bf16* bcol = wos + (size_t)lm * WOS_STRIDE;
      const __bf16* hrow = hst + (size_t)pnxt * BB * HID + (size_t)(r_mt * 16 + lm) * HID;
      const __bf16* rrow = rst + (size_t)pcur * BB * OUT + (size_t)(r_mt * 16 + lm) * OUT;
      const int kbase = r_hi ? 12 : 0;
#pragma unroll 4
      for (int j = 0; j < 12; ++j) {
        int k = (kbase + j) * 32;
        const __bf16* arow = (k < HID) ? (hrow + k) : (rrow + (k - HID));
        v16bf a = make_a_bf(arow, half);
        v16bf b = make_b(bcol + k, half);
        c = wmma_bf16(a, b, c);
      }
      if (r_hi) {
#pragma unroll
        for (int i = 0; i < 8; ++i) part[r_mt][i + half * 8][lm] = c[i];
      }
      __syncthreads();
      if (!r_hi) {
#pragma unroll
        for (int i = 0; i < 8; ++i) {
          float v = c[i] + part[r_mt][i + half * 8][lm];
          v = v > 0.f ? v : 0.f;
          int brow = r_mt * 16 + i + half * 8;
          rout[((size_t)brow * 1025 + (t + 1)) * OUT + r_o] = v;
          rst[(size_t)pnxt * BB * OUT + (size_t)brow * OUT + r_o] = (__bf16)v;
        }
      }
    }
  }
}

// ---------------------------------------------------------------------------
// Launch
// ---------------------------------------------------------------------------
extern "C" void kernel_launch(void* const* d_in, const int* in_sizes, int n_in,
                              void* d_out, int out_size, void* d_ws, size_t ws_size,
                              hipStream_t stream) {
  (void)in_sizes; (void)n_in; (void)out_size; (void)ws_size;
  const float* x    = (const float*)d_in[0];
  const float* h0   = (const float*)d_in[1];
  const float* r0   = (const float*)d_in[2];
  const float* wih  = (const float*)d_in[3];
  const float* b_ih = (const float*)d_in[4];
  const float* who  = (const float*)d_in[5];
  const float* b_ho = (const float*)d_in[6];

  float* rout = (float*)d_out;                                 // [64, 1025, 256]
  float* hout = rout + (size_t)BB * 1025 * OUT;                // [64, 1025, 512]

  unsigned* bar   = (unsigned*)((char*)d_ws + WS_BAR);
  __bf16* wih_bf  = (__bf16*)((char*)d_ws + WS_WIHBF);
  __bf16* hst     = (__bf16*)((char*)d_ws + WS_HST);
  __bf16* rst     = (__bf16*)((char*)d_ws + WS_RST);

  hipMemsetAsync(d_ws, 0, 256, stream);                        // reset barrier

  {
    int n = HID * WIH_K;                                       // 393216 dominant
    setup_kernel<<<(n + 255) / 256, 256, 0, stream>>>(wih, wih_bf, h0, r0,
                                                      rout, hout, hst, rst);
  }
  {
    int waves = (BB * TT / 16) * (HID / 16);                   // 131072 tiles
    xw_kernel<<<waves / 8, 256, 0, stream>>>(x, wih_bf, b_ih, hout);
  }
  rnn_kernel<<<NWG, 256, 0, stream>>>(wih, who, b_ho, rout, hout, hst, rst, bar);
}
